// Gnn_59734405152882
// MI455X (gfx1250) — matrix operand order repaired and checked by reading, not compiled, not run on hardware
//
#include <hip/hip_runtime.h>

typedef _Float16 v16h __attribute__((ext_vector_type(16)));
typedef _Float16 v8h  __attribute__((ext_vector_type(8)));
typedef float    v8f  __attribute__((ext_vector_type(8)));
typedef float    v4f  __attribute__((ext_vector_type(4)));

#define DEVINL __device__ __forceinline__
#define WB() __builtin_amdgcn_wave_barrier()

// ---------------------------------------------------------------------------
// WMMA fragment helpers (CDNA5 f16 16x16x32, wave32 layouts per ISA 7.12.2)
// A (16x32): lane holds row = lane&15; halves[0..7] = K hi*8..hi*8+7,
//            halves[8..15] = K 16+hi*8.. (hi = lane>>4), +kt*32.
// B (32x16): lane holds col = lane&15; halves[0..7] = K hi*8+h, halves[8..15] = K 16+hi*8+(h-8) (same as A).
// C/D (16x16 f32): lane holds col N = lane&15; vgpr v holds row M = v + 8*(lane>>4).
// ---------------------------------------------------------------------------

DEVINL v16h make_v16(v8h lo, v8h hi2) {
  union { v16h v; v8h h[2]; } u;
  u.h[0] = lo; u.h[1] = hi2;
  return u.v;
}

DEVINL v8h cvt8(v4f a, v4f b) {
  v8h r;
#pragma unroll
  for (int j = 0; j < 4; ++j) { r[j] = (_Float16)a[j]; r[4 + j] = (_Float16)b[j]; }
  return r;
}

// Build A fragment from 32 consecutive f32 features of this lane's row.
DEVINL v16h a_from_g32(const float* seg, int lane) {
  int hb = (lane >> 4) * 8;
  v4f a0 = *(const v4f*)(seg + hb);
  v4f a1 = *(const v4f*)(seg + hb + 4);
  v4f b0 = *(const v4f*)(seg + 16 + hb);
  v4f b1 = *(const v4f*)(seg + 16 + hb + 4);
  return make_v16(cvt8(a0, a1), cvt8(b0, b1));
}

// Build A fragment from LDS h-stage laid out [16 rows][64 cols] f16.
DEVINL v16h a_from_lds(const _Float16* hs, int kt, int lane) {
  int row = lane & 15, hb = (lane >> 4) * 8;
  const _Float16* p = hs + row * 64 + kt * 32 + hb;
  return make_v16(*(const v8h*)p, *(const v8h*)(p + 16));
}

// Zero-padded A fragment for tiny K (encoders: K=6 or 4, row stride = din).
DEVINL v16h a_pad(const float* rowp, int din, int lane) {
  int hb = (lane >> 4) * 8;
  v16h r;
#pragma unroll
  for (int j = 0; j < 8; ++j) {
    int K0 = hb + j, K1 = 16 + hb + j;
    r[j]     = (K0 < din) ? (_Float16)rowp[K0] : (_Float16)0.f;
    r[8 + j] = (K1 < din) ? (_Float16)rowp[K1] : (_Float16)0.f;
  }
  return r;
}

template <int Np>
DEVINL void mma_nt(const _Float16* wl, int kt, int lane, v16h a, v8f* acc) {
#pragma unroll
  for (int nt = 0; nt < Np; ++nt) {
    v16h b = *(const v16h*)(wl + ((kt * Np + nt) * 32 + lane) * 16);
    acc[nt] = __builtin_amdgcn_wmma_f32_16x16x32_f16(false, a, false, b,
                                                     (short)0, acc[nt], false, false);
  }
}

template <int Np>
DEVINL void init_bias(v8f* acc, const float* bias, int lane) {
  int col = lane & 15;
#pragma unroll
  for (int nt = 0; nt < Np; ++nt) {
    float bv = bias[nt * 16 + col];
    v8f c;
#pragma unroll
    for (int v = 0; v < 8; ++v) c[v] = bv;
    acc[nt] = c;
  }
}

DEVINL void stage_store_h(_Float16* hs, const v8f* acc, int lane, bool dorelu) {
  int col = lane & 15, mb = (lane >> 4) * 8;
#pragma unroll
  for (int nt = 0; nt < 4; ++nt)
#pragma unroll
    for (int v = 0; v < 8; ++v) {
      float x = acc[nt][v];
      if (dorelu) x = fmaxf(x, 0.f);
      hs[(v + mb) * 64 + nt * 16 + col] = (_Float16)x;
    }
}

DEVINL void stage_store_f(float* fs, const v8f* acc, int lane) {
  int col = lane & 15, mb = (lane >> 4) * 8;
#pragma unroll
  for (int nt = 0; nt < 4; ++nt)
#pragma unroll
    for (int v = 0; v < 8; ++v)
      fs[(v + mb) * 64 + nt * 16 + col] = acc[nt][v];
}

DEVINL void hidden_layer(const _Float16* hs, const _Float16* wl,
                         const float* bias, v8f* acc, int lane) {
  init_bias<4>(acc, bias, lane);
#pragma unroll
  for (int kt = 0; kt < 2; ++kt)
    mma_nt<4>(wl, kt, lane, a_from_lds(hs, kt, lane), acc);
}

// ---------------------------------------------------------------------------
// Weight packer: f32 [Korig,Norig] row-major -> f16 B-fragment layout,
// K padded to Kp*32, N padded to Np*16.
// packed[((kt*Np+nt)*32 + lane)*16 + h] = W[(kt*32+((h&8)<<1)+(lane>>4)*8+(h&7))*Norig + nt*16+(lane&15)]
// ---------------------------------------------------------------------------
__global__ void pack_w(const float* __restrict__ W, _Float16* __restrict__ dst,
                       int Korig, int Norig, int Kp, int Np) {
  int idx = blockIdx.x * blockDim.x + threadIdx.x;
  int total = Kp * Np * 512;
  if (idx >= total) return;
  int kt = idx / (Np * 512);
  int rem = idx - kt * (Np * 512);
  int nt = rem >> 9;
  int li = rem & 511;
  int lane = li >> 4, h = li & 15;
  int col = nt * 16 + (lane & 15);
  int K = kt * 32 + ((h & 8) << 1) + (lane >> 4) * 8 + (h & 7);
  float v = (K < Korig && col < Norig) ? W[K * Norig + col] : 0.f;
  dst[idx] = (_Float16)v;
}

__global__ void zero_kernel(float* __restrict__ p, int n) {
  int i = blockIdx.x * blockDim.x + threadIdx.x;
  if (i < n) p[i] = 0.f;
}

// ---------------------------------------------------------------------------
// Encoder: rows x din -> rows x 64 (3-layer MLP, no residual)
// ---------------------------------------------------------------------------
__global__ void __launch_bounds__(256) encode_kernel(
    const float* __restrict__ in, float* __restrict__ outb,
    const _Float16* __restrict__ wpack,
    const float* __restrict__ b1, const float* __restrict__ b2,
    const float* __restrict__ b3, int ntiles, int din) {
  __shared__ __align__(32) _Float16 wl[10240];
  __shared__ float bl[192];
  __shared__ __align__(16) float fst[8][1024];
  for (int i = threadIdx.x; i < 10240 / 8; i += blockDim.x)
    *(v8h*)&wl[i * 8] = *(const v8h*)&wpack[i * 8];
  for (int i = threadIdx.x; i < 64; i += blockDim.x) {
    bl[i] = b1[i]; bl[64 + i] = b2[i]; bl[128 + i] = b3[i];
  }
  __syncthreads();
  int lane = threadIdx.x & 31, wv = threadIdx.x >> 5;
  float* fs = fst[wv];
  _Float16* hs = (_Float16*)fs;
  int gw = blockIdx.x * 8 + wv, nw = gridDim.x * 8;
  for (int t = gw; t < ntiles; t += nw) {
    int row = t * 16 + (lane & 15);
    v8f acc[4];
    init_bias<4>(acc, bl, lane);
    mma_nt<4>(wl, 0, lane, a_pad(in + (size_t)row * din, din, lane), acc);
    WB(); stage_store_h(hs, acc, lane, true); WB();
    hidden_layer(hs, wl + 2048, bl + 64, acc, lane);
    WB(); stage_store_h(hs, acc, lane, true); WB();
    hidden_layer(hs, wl + 6144, bl + 128, acc, lane);
    WB(); stage_store_f(fs, acc, lane); WB();
    float* op = outb + (size_t)t * 1024;
#pragma unroll
    for (int j = 0; j < 8; ++j) {
      int q = (j * 32 + lane) * 4;
      *(v4f*)&op[q] = *(const v4f*)&fs[q];
    }
  }
}

// ---------------------------------------------------------------------------
// Edge processor step: eh = (MLP([eh, nh[src], nh[dst]]) + eh) * 0.5
// plus atomic segment-sum of new eh into esum[src].
// ---------------------------------------------------------------------------
__global__ void __launch_bounds__(256) edge_step_kernel(
    const int* __restrict__ edges, float* __restrict__ eh,
    const float* __restrict__ nh, float* __restrict__ esum,
    const _Float16* __restrict__ wpack,
    const float* __restrict__ b1, const float* __restrict__ b2,
    const float* __restrict__ b3, int ntiles, int tilesPerBatch, int nodesPerBatch) {
  __shared__ __align__(32) _Float16 wl[20480];
  __shared__ float bl[192];
  __shared__ __align__(16) float fst[8][1024];
  __shared__ int srcl[8][16];
  for (int i = threadIdx.x; i < 20480 / 8; i += blockDim.x)
    *(v8h*)&wl[i * 8] = *(const v8h*)&wpack[i * 8];
  for (int i = threadIdx.x; i < 64; i += blockDim.x) {
    bl[i] = b1[i]; bl[64 + i] = b2[i]; bl[128 + i] = b3[i];
  }
  __syncthreads();
  int lane = threadIdx.x & 31, wv = threadIdx.x >> 5;
  float* fs = fst[wv];
  _Float16* hs = (_Float16*)fs;
  int gw = blockIdx.x * 8 + wv, nw = gridDim.x * 8;
  for (int t = gw; t < ntiles; t += nw) {
    int b = t / tilesPerBatch;
    int row = t * 16 + (lane & 15);
    int s = edges[(size_t)row * 2] + b * nodesPerBatch;
    int d = edges[(size_t)row * 2 + 1] + b * nodesPerBatch;
    if (lane < 16) srcl[wv][lane] = s;
    const float* er = eh + (size_t)row * 64;
    const float* sr = nh + (size_t)s * 64;
    const float* dr = nh + (size_t)d * 64;
    v8f acc[4];
    init_bias<4>(acc, bl, lane);
    mma_nt<4>(wl, 0, lane, a_from_g32(er, lane), acc);
    mma_nt<4>(wl, 1, lane, a_from_g32(er + 32, lane), acc);
    mma_nt<4>(wl, 2, lane, a_from_g32(sr, lane), acc);
    mma_nt<4>(wl, 3, lane, a_from_g32(sr + 32, lane), acc);
    mma_nt<4>(wl, 4, lane, a_from_g32(dr, lane), acc);
    mma_nt<4>(wl, 5, lane, a_from_g32(dr + 32, lane), acc);
    WB(); stage_store_h(hs, acc, lane, true); WB();
    hidden_layer(hs, wl + 12288, bl + 64, acc, lane);
    WB(); stage_store_h(hs, acc, lane, true); WB();
    hidden_layer(hs, wl + 16384, bl + 128, acc, lane);
    WB(); stage_store_f(fs, acc, lane); WB();
    float* ew = eh + (size_t)t * 1024;
#pragma unroll
    for (int j = 0; j < 8; ++j) {
      int q = j * 32 + lane;
      v4f nv = *(const v4f*)&fs[q * 4];
      v4f ov = *(const v4f*)&ew[q * 4];
      nv = (nv + ov) * 0.5f;
      *(v4f*)&ew[q * 4] = nv;
      int rl = q >> 4, c0 = (q & 15) * 4;
      float* epx = esum + (size_t)srcl[wv][rl] * 64 + c0;
      __hip_atomic_fetch_add(epx + 0, nv.x, __ATOMIC_RELAXED, __HIP_MEMORY_SCOPE_AGENT);
      __hip_atomic_fetch_add(epx + 1, nv.y, __ATOMIC_RELAXED, __HIP_MEMORY_SCOPE_AGENT);
      __hip_atomic_fetch_add(epx + 2, nv.z, __ATOMIC_RELAXED, __HIP_MEMORY_SCOPE_AGENT);
      __hip_atomic_fetch_add(epx + 3, nv.w, __ATOMIC_RELAXED, __HIP_MEMORY_SCOPE_AGENT);
    }
  }
}

// ---------------------------------------------------------------------------
// Node processor step: nh = (MLP([nh, esum]) + nh) * 0.5
// ---------------------------------------------------------------------------
__global__ void __launch_bounds__(256) node_step_kernel(
    float* __restrict__ nh, const float* __restrict__ esum,
    const _Float16* __restrict__ wpack,
    const float* __restrict__ b1, const float* __restrict__ b2,
    const float* __restrict__ b3, int ntiles) {
  __shared__ __align__(32) _Float16 wl[16384];
  __shared__ float bl[192];
  __shared__ __align__(16) float fst[8][1024];
  for (int i = threadIdx.x; i < 16384 / 8; i += blockDim.x)
    *(v8h*)&wl[i * 8] = *(const v8h*)&wpack[i * 8];
  for (int i = threadIdx.x; i < 64; i += blockDim.x) {
    bl[i] = b1[i]; bl[64 + i] = b2[i]; bl[128 + i] = b3[i];
  }
  __syncthreads();
  int lane = threadIdx.x & 31, wv = threadIdx.x >> 5;
  float* fs = fst[wv];
  _Float16* hs = (_Float16*)fs;
  int gw = blockIdx.x * 8 + wv, nw = gridDim.x * 8;
  for (int t = gw; t < ntiles; t += nw) {
    int row = t * 16 + (lane & 15);
    const float* xr = nh + (size_t)row * 64;
    const float* er = esum + (size_t)row * 64;
    v8f acc[4];
    init_bias<4>(acc, bl, lane);
    mma_nt<4>(wl, 0, lane, a_from_g32(xr, lane), acc);
    mma_nt<4>(wl, 1, lane, a_from_g32(xr + 32, lane), acc);
    mma_nt<4>(wl, 2, lane, a_from_g32(er, lane), acc);
    mma_nt<4>(wl, 3, lane, a_from_g32(er + 32, lane), acc);
    WB(); stage_store_h(hs, acc, lane, true); WB();
    hidden_layer(hs, wl + 8192, bl + 64, acc, lane);
    WB(); stage_store_h(hs, acc, lane, true); WB();
    hidden_layer(hs, wl + 12288, bl + 128, acc, lane);
    WB(); stage_store_f(fs, acc, lane); WB();
    float* ow = nh + (size_t)t * 1024;
#pragma unroll
    for (int j = 0; j < 8; ++j) {
      int q = (j * 32 + lane) * 4;
      v4f nv = (*(const v4f*)&fs[q] + *(const v4f*)&ow[q]) * 0.5f;
      *(v4f*)&ow[q] = nv;
    }
  }
}

// ---------------------------------------------------------------------------
// Decoder: out = n_feats + MLP_dec(nh), last layer 64 -> 6 (N padded to 16)
// ---------------------------------------------------------------------------
__global__ void __launch_bounds__(256) decode_kernel(
    const float* __restrict__ nh, const float* __restrict__ nfeat,
    float* __restrict__ outp, const _Float16* __restrict__ wpack,
    const float* __restrict__ b1, const float* __restrict__ b2,
    const float* __restrict__ b3, int ntiles) {
  __shared__ __align__(32) _Float16 wl[9216];
  __shared__ float bl[144];
  __shared__ __align__(16) float fst[8][1024];
  for (int i = threadIdx.x; i < 9216 / 8; i += blockDim.x)
    *(v8h*)&wl[i * 8] = *(const v8h*)&wpack[i * 8];
  for (int i = threadIdx.x; i < 64; i += blockDim.x) {
    bl[i] = b1[i]; bl[64 + i] = b2[i];
  }
  for (int i = threadIdx.x; i < 16; i += blockDim.x) bl[128 + i] = (i < 6) ? b3[i] : 0.f;
  __syncthreads();
  int lane = threadIdx.x & 31, wv = threadIdx.x >> 5;
  float* fs = fst[wv];
  _Float16* hs = (_Float16*)fs;
  int gw = blockIdx.x * 8 + wv, nw = gridDim.x * 8;
  for (int t = gw; t < ntiles; t += nw) {
    const float* xr = nh + (size_t)(t * 16 + (lane & 15)) * 64;
    v8f acc[4];
    init_bias<4>(acc, bl, lane);
    mma_nt<4>(wl, 0, lane, a_from_g32(xr, lane), acc);
    mma_nt<4>(wl, 1, lane, a_from_g32(xr + 32, lane), acc);
    WB(); stage_store_h(hs, acc, lane, true); WB();
    hidden_layer(hs, wl + 4096, bl + 64, acc, lane);
    WB(); stage_store_h(hs, acc, lane, true); WB();
    v8f acc1[1];
    init_bias<1>(acc1, bl + 128, lane);
#pragma unroll
    for (int kt = 0; kt < 2; ++kt)
      mma_nt<1>(wl + 8192, kt, lane, a_from_lds(hs, kt, lane), acc1);
    WB();
    {
      int col = lane & 15, mb = (lane >> 4) * 8;
#pragma unroll
      for (int v = 0; v < 8; ++v) fs[(v + mb) * 16 + col] = acc1[0][v];
    }
    WB();
#pragma unroll
    for (int j = 0; j < 3; ++j) {
      int q = j * 32 + lane;
      if (q < 96) {
        int r = q / 6, c = q - r * 6;
        size_t gi = (size_t)(t * 16 + r) * 6 + c;
        outp[gi] = nfeat[gi] + fs[r * 16 + c];
      }
    }
  }
}

// ---------------------------------------------------------------------------
// Host launcher
// ---------------------------------------------------------------------------
extern "C" void kernel_launch(void* const* d_in, const int* in_sizes, int n_in,
                              void* d_out, int out_size, void* d_ws, size_t ws_size,
                              hipStream_t stream) {
  (void)out_size; (void)ws_size;
  const int Bc = 2, Nn = 20000, Ee = 320000, NFc = 6, EFc = 4;

  const int* edges = nullptr;
  const float* nfeats = nullptr;
  const float* efeats = nullptr;
  int pidx[128]; int np = 0;
  for (int i = 0; i < n_in; ++i) {
    long s = in_sizes[i];
    if (s == (long)Bc * Ee * 2) edges = (const int*)d_in[i];
    else if (s == (long)Bc * Nn * NFc) nfeats = (const float*)d_in[i];
    else if (s == (long)Bc * Ee * EFc) efeats = (const float*)d_in[i];
    else if (np < 128) pidx[np++] = i;
  }

  struct HB { const float *w1, *b1, *w2, *b2, *w3, *b3; };
  HB n_enc{}, e_enc{}, dec{}, e_proc[4]{}, n_proc[4]{};
  int ep = 0, npn = 0;
  // Params arrive as 11 blocks of 6 leaves each; handle either leaf order:
  // insertion (w1,b1,w2,b2,w3,b3) or sorted-key (b1,b2,b3,w1,w2,w3).
  for (int k = 0; k + 5 < np; k += 6) {
    const float* lf[6]; int ls[6];
    for (int j = 0; j < 6; ++j) {
      lf[j] = (const float*)d_in[pidx[k + j]];
      ls[j] = in_sizes[pidx[k + j]];
    }
    HB blk; int w1sz;
    if (ls[0] <= 64) { blk = {lf[3], lf[0], lf[4], lf[1], lf[5], lf[2]}; w1sz = ls[3]; }
    else             { blk = {lf[0], lf[1], lf[2], lf[3], lf[4], lf[5]}; w1sz = ls[0]; }
    bool has6 = false;
    for (int j = 0; j < 6; ++j) if (ls[j] == 6) has6 = true;
    if (has6) dec = blk;
    else if (w1sz == 256) e_enc = blk;
    else if (w1sz == 12288) { if (ep < 4) e_proc[ep++] = blk; }
    else if (w1sz == 8192)  { if (npn < 4) n_proc[npn++] = blk; }
    else n_enc = blk; // w1sz == 384
  }

  // Workspace layout (bytes): packed f16 weights | nh | esum | eh  (~185 MB)
  _Float16* pk = (_Float16*)d_ws;
  char* wsb = (char*)d_ws;
  float* nh   = (float*)(wsb + 354304);
  float* esum = (float*)(wsb + 10594304);
  float* ehb  = (float*)(wsb + 20834304);

  auto pw = [&](const float* W, long off, int Ko, int No, int Kp, int Npt) {
    int total = Kp * Npt * 512;
    pack_w<<<(total + 255) / 256, 256, 0, stream>>>(W, pk + off, Ko, No, Kp, Npt);
  };
  pw(n_enc.w1, 0, 6, 64, 1, 4);      pw(n_enc.w2, 2048, 64, 64, 2, 4);  pw(n_enc.w3, 6144, 64, 64, 2, 4);
  pw(e_enc.w1, 10240, 4, 64, 1, 4);  pw(e_enc.w2, 12288, 64, 64, 2, 4); pw(e_enc.w3, 16384, 64, 64, 2, 4);
  for (int i = 0; i < 4; ++i) {
    long b0 = 20480 + (long)i * 20480;
    pw(e_proc[i].w1, b0, 192, 64, 6, 4);
    pw(e_proc[i].w2, b0 + 12288, 64, 64, 2, 4);
    pw(e_proc[i].w3, b0 + 16384, 64, 64, 2, 4);
  }
  for (int i = 0; i < 4; ++i) {
    long b0 = 102400 + (long)i * 16384;
    pw(n_proc[i].w1, b0, 128, 64, 4, 4);
    pw(n_proc[i].w2, b0 + 8192, 64, 64, 2, 4);
    pw(n_proc[i].w3, b0 + 12288, 64, 64, 2, 4);
  }
  pw(dec.w1, 167936, 64, 64, 2, 4);  pw(dec.w2, 172032, 64, 64, 2, 4);  pw(dec.w3, 176128, 64, 6, 2, 1);

  const int ntilesN = Bc * Nn / 16;   // 2500
  const int ntilesE = Bc * Ee / 16;   // 40000
  encode_kernel<<<160, 256, 0, stream>>>(nfeats, nh, pk, n_enc.b1, n_enc.b2, n_enc.b3, ntilesN, NFc);
  encode_kernel<<<640, 256, 0, stream>>>(efeats, ehb, pk + 10240, e_enc.b1, e_enc.b2, e_enc.b3, ntilesE, EFc);

  const int esumN = Bc * Nn * 64;     // 2,560,000 f32
  for (int p = 0; p < 4; ++p) {
    zero_kernel<<<(esumN + 255) / 256, 256, 0, stream>>>(esum, esumN);
    edge_step_kernel<<<640, 256, 0, stream>>>(edges, ehb, nh, esum,
        pk + 20480 + (long)p * 20480, e_proc[p].b1, e_proc[p].b2, e_proc[p].b3,
        ntilesE, Ee / 16, Nn);
    node_step_kernel<<<160, 256, 0, stream>>>(nh, esum,
        pk + 102400 + (long)p * 16384, n_proc[p].b1, n_proc[p].b2, n_proc[p].b3, ntilesN);
  }
  decode_kernel<<<160, 256, 0, stream>>>(nh, nfeats, (float*)d_out,
      pk + 167936, dec.b1, dec.b2, dec.b3, ntilesN);
}
